// RelationalCritic_7980049236588
// MI455X (gfx1250) — matrix operand order repaired and checked by reading, not compiled, not run on hardware
//
#include <hip/hip_runtime.h>

// ---------------------------------------------------------------------------
// CDNA5 (gfx1250) wave32 WMMA implementation of the relational critic.
// ---------------------------------------------------------------------------

typedef __bf16 bf16_t;
typedef __attribute__((ext_vector_type(16))) __bf16 v16bf;
typedef __attribute__((ext_vector_type(8)))  float  v8f;

#define B_    2048
#define N_    32
#define F_    64
#define E_    8
#define H_    128
#define NA_   4
#define NACT_ 16

// ---- float -> bf16 via native converter (v_cvt_bf16_f32 / v_cvt_pk_...) --
__device__ __forceinline__ bf16_t f2bf(float f) {
  return (bf16_t)f;   // clang lowers to hardware RNE convert on gfx1250
}

// ---- WMMA fragment loaders (ISA 7.12.2 layouts, wave32) -----------------
// A (16x32 bf16, MxK): lane L -> m=L%16, half=L/16;
//   components t<8 : K = half*8 + t ; t>=8 : K = 16 + half*8 + (t-8)
__device__ __forceinline__ v16bf frag_a(const bf16_t* p, int stride, int lane) {
  const int m = lane & 15, h = lane >> 4;
  const bf16_t* r = p + m * stride + h * 8;
  v16bf f;
#pragma unroll
  for (int t = 0; t < 8; ++t) f[t] = r[t];
#pragma unroll
  for (int t = 0; t < 8; ++t) f[8 + t] = r[16 + t];
  return f;
}

// B (32x16 bf16, KxN) loaded from a B^T (row-major [n][k]) buffer:
// lane L -> n=L%16, half=L/16; t<8 : K = half*8 + t ; t>=8 : K = 16 + half*8 + (t-8)
__device__ __forceinline__ v16bf frag_b(const bf16_t* p, int stride, int lane) {
  const int n = lane & 15, h = lane >> 4;
  const bf16_t* r = p + n * stride + h * 8;
  v16bf f;
#pragma unroll
  for (int t = 0; t < 16; ++t) f[t] = r[t < 8 ? t : t + 8];
  return f;
}

__device__ __forceinline__ v8f wmma_bf16(v16bf a, v16bf b, v8f c) {
  // D = A*B + C, f32 accumulate (probe-confirmed builtin, 8 args)
  return __builtin_amdgcn_wmma_f32_16x16x32_bf16(false, a, false, b,
                                                 (short)0, c, false, false);
}

// ---------------------------------------------------------------------------
// Kernel 0: convert weights to bf16 B^T layouts in workspace.
//   WrelT [E][H][H] : WrelT[e][n][k] = Wrel[e][k][n]      (131072 elems)
//   WrootT[H][H]    : WrootT[n][k]   = Wroot[k][n]        ( 16384)
//   WeT   [H][F]    : WeT[n][k]      = We[k][n]           (  8192)
//   W1T [NA][H][H]  : W1T[a][n][k]   = W1[a][k][n]        ( 65536)
//   W2T [NA][NACT][H]: W2T[a][n][k]  = W2[a][k][n]        (  8192)
// ---------------------------------------------------------------------------
__global__ void relcrit_prep(const float* __restrict__ We,
                             const float* __restrict__ Wrel,
                             const float* __restrict__ Wroot,
                             const float* __restrict__ W1,
                             const float* __restrict__ W2,
                             bf16_t* __restrict__ WrelT,
                             bf16_t* __restrict__ WrootT,
                             bf16_t* __restrict__ WeT,
                             bf16_t* __restrict__ W1T,
                             bf16_t* __restrict__ W2T) {
  const int i = blockIdx.x * 256 + threadIdx.x;
  if (i < 131072) {
    int e = i >> 14, r = i & 16383, n = r >> 7, k = r & 127;
    WrelT[i] = f2bf(Wrel[e * 16384 + k * 128 + n]);
  } else if (i < 147456) {
    int j = i - 131072, n = j >> 7, k = j & 127;
    WrootT[j] = f2bf(Wroot[k * 128 + n]);
  } else if (i < 155648) {
    int j = i - 147456, n = j >> 6, k = j & 63;
    WeT[j] = f2bf(We[k * 128 + n]);
  } else if (i < 221184) {
    int j = i - 155648, a = j >> 14, r = j & 16383, n = r >> 7, k = r & 127;
    W1T[j] = f2bf(W1[a * 16384 + k * 128 + n]);
  } else if (i < 229376) {
    int j = i - 221184, a = j >> 11, r = j & 2047, n = r >> 7, k = r & 127;
    W2T[j] = f2bf(W2[a * 2048 + k * 16 + n]);
  }
}

// ---------------------------------------------------------------------------
// Kernel 1: per-batch relational block (one workgroup = 8 wave32 per b).
//   x = unary@We+be -> y_e = Anorm_e^T @ x -> h = sum_e y_e@Wrel_e + x@Wroot
//   + brel -> relu -> max over nodes -> pooled[b] (bf16)
// Wave w owns output column tile n = w (16 of 128 columns) for both 16-row
// M-tiles; h accumulators live in v8f registers across all 36 K-steps.
// ---------------------------------------------------------------------------
__global__ void __launch_bounds__(256)
relcrit_main(const float* __restrict__ unary,
             const int* __restrict__ binary,
             const float* __restrict__ be,
             const float* __restrict__ brel,
             const bf16_t* __restrict__ WeT,
             const bf16_t* __restrict__ WrelT,
             const bf16_t* __restrict__ WrootT,
             bf16_t* __restrict__ pooled) {
  __shared__ bf16_t sU[N_][72];        // unary[b] (A-operand, K=F)
  __shared__ bf16_t sXA[N_][136];      // x row-major (A-operand, K=H)
  __shared__ bf16_t sXT[H_][40];       // x^T (B^T operand for y-phase)
  __shared__ bf16_t sAT[E_][N_][40];   // Anorm^T[e][j][i]
  __shared__ bf16_t sY[N_][136];       // y_e (A-operand, K=H)
  __shared__ float  sPool[2][H_];

  const int tid  = threadIdx.x;
  const int lane = tid & 31;
  const int w    = tid >> 5;
  const int nl   = lane & 15;
  const int hh   = lane >> 4;
  const int b    = blockIdx.x;

  // Warm the shared weight tables in cache (they live in L2; 192MB).
  {
    const char* wp = (const char*)WrelT;
    for (int off = tid * 64; off < E_ * H_ * H_ * 2; off += 256 * 64 * 4)
      __builtin_prefetch(wp + off, 0, 1);
  }

  // --- stage unary[b] as bf16 (coalesced float4 loads) -------------------
  {
    const float4* pu = (const float4*)(unary + (size_t)b * (N_ * F_));
    float4 v0 = pu[tid * 2], v1 = pu[tid * 2 + 1];
    int base = tid * 8, i = base >> 6, f0 = base & 63;
    sU[i][f0 + 0] = f2bf(v0.x); sU[i][f0 + 1] = f2bf(v0.y);
    sU[i][f0 + 2] = f2bf(v0.z); sU[i][f0 + 3] = f2bf(v0.w);
    sU[i][f0 + 4] = f2bf(v1.x); sU[i][f0 + 5] = f2bf(v1.y);
    sU[i][f0 + 6] = f2bf(v1.z); sU[i][f0 + 7] = f2bf(v1.w);
  }

  // --- mean-normalized adjacency transpose: sAT[e][j][i] = A[b,i,j,e]/deg -
  {
    const int e = tid >> 5, j = tid & 31;
    const int* pb = binary + (((size_t)b * N_) * N_ + j) * E_ + e;  // i stride N_*E_
    int deg = 0;
#pragma unroll 8
    for (int i = 0; i < N_; ++i) deg += pb[i * (N_ * E_)];
    float scale = deg > 0 ? 1.0f / (float)deg : 0.0f;
    bf16_t sb = f2bf(scale), zb = f2bf(0.0f);
#pragma unroll 8
    for (int i = 0; i < N_; ++i) sAT[e][j][i] = pb[i * (N_ * E_)] ? sb : zb;
  }
  __syncthreads();

  // --- Phase A: x = unary@We + be  (M=32, K=64, N=128) -------------------
#pragma unroll
  for (int m = 0; m < 2; ++m) {
    v8f acc = {};
#pragma unroll
    for (int kt = 0; kt < 2; ++kt) {
      v16bf a  = frag_a(&sU[m * 16][kt * 32], 72, lane);
      v16bf bb = frag_b(WeT + (w * 16) * F_ + kt * 32, F_, lane);
      acc = wmma_bf16(a, bb, acc);
    }
    const int col = w * 16 + nl;
    const float bias = be[col];
#pragma unroll
    for (int v = 0; v < 8; ++v) {
      int row = m * 16 + hh * 8 + v;
      bf16_t bv = f2bf(acc[v] + bias);
      sXA[row][col] = bv;   // A-operand layout
      sXT[col][row] = bv;   // B^T layout for the aggregation GEMM
    }
  }
  __syncthreads();

  // --- root-weight contribution: h += x@Wroot ----------------------------
  v8f accH0 = {}, accH1 = {};
#pragma unroll
  for (int kt = 0; kt < 4; ++kt) {
    v16bf bb = frag_b(WrootT + (w * 16) * H_ + kt * 32, H_, lane);
    v16bf a0 = frag_a(&sXA[0][kt * 32], 136, lane);
    v16bf a1 = frag_a(&sXA[16][kt * 32], 136, lane);
    accH0 = wmma_bf16(a0, bb, accH0);
    accH1 = wmma_bf16(a1, bb, accH1);
  }

  // --- per-relation: y_e = Anorm_e^T @ x ; h += y_e @ Wrel_e -------------
  for (int e = 0; e < E_; ++e) {
#pragma unroll
    for (int m = 0; m < 2; ++m) {
      v16bf a  = frag_a(&sAT[e][m * 16][0], 40, lane);   // M=j, K=i
      v16bf bb = frag_b(&sXT[w * 16][0], 40, lane);      // K=i, N=d
      v8f acc = {};
      acc = wmma_bf16(a, bb, acc);
#pragma unroll
      for (int v = 0; v < 8; ++v)
        sY[m * 16 + hh * 8 + v][w * 16 + nl] = f2bf(acc[v]);
    }
    __syncthreads();  // all columns of y_e visible
#pragma unroll
    for (int kt = 0; kt < 4; ++kt) {
      v16bf bb = frag_b(WrelT + e * (H_ * H_) + (w * 16) * H_ + kt * 32, H_, lane);
      v16bf a0 = frag_a(&sY[0][kt * 32], 136, lane);
      v16bf a1 = frag_a(&sY[16][kt * 32], 136, lane);
      accH0 = wmma_bf16(a0, bb, accH0);
      accH1 = wmma_bf16(a1, bb, accH1);
    }
    __syncthreads();  // before next e overwrites sY
  }

  // --- +brel, relu, max-pool over the 32 nodes ---------------------------
  {
    const int col = w * 16 + nl;
    const float bias = brel[col];
    float m8 = 0.0f;  // relu => pooled >= 0
#pragma unroll
    for (int v = 0; v < 8; ++v) {
      m8 = fmaxf(m8, accH0[v] + bias);
      m8 = fmaxf(m8, accH1[v] + bias);
    }
    sPool[hh][col] = m8;
  }
  __syncthreads();
  if (tid < H_) {
    float p = fmaxf(sPool[0][tid], sPool[1][tid]);
    pooled[(size_t)b * H_ + tid] = f2bf(p);
  }
}

// ---------------------------------------------------------------------------
// Kernel 2: per-agent heads. 64 blocks x 8 waves; each wave owns 16 rows of
// one agent: z = pooled@W1[a]+b1 -> leaky-relu -> all_q = z@W2[a]+b2 ->
// gather at argmax(actions).
// ---------------------------------------------------------------------------
__global__ void __launch_bounds__(256)
relcrit_heads(const bf16_t* __restrict__ pooled,
              const bf16_t* __restrict__ W1T,
              const bf16_t* __restrict__ W2T,
              const float* __restrict__ b1,
              const float* __restrict__ b2,
              const float* __restrict__ actions,
              float* __restrict__ qout) {
  __shared__ bf16_t sZ[8][16][136];
  __shared__ float  sQ[8][16][17];

  const int tid  = threadIdx.x;
  const int lane = tid & 31;
  const int w    = tid >> 5;
  const int nl   = lane & 15;
  const int hh   = lane >> 4;
  const int a    = blockIdx.x >> 4;                       // 16 blocks / agent
  const int r0   = ((blockIdx.x & 15) * 8 + w) * 16;      // 16 rows / wave

  // cache the 4 A-fragments of pooled rows (K=128)
  v16bf af[4];
#pragma unroll
  for (int kt = 0; kt < 4; ++kt)
    af[kt] = frag_a(pooled + (size_t)r0 * H_ + kt * 32, H_, lane);

  const bf16_t* w1 = W1T + a * (H_ * H_);
#pragma unroll
  for (int nt = 0; nt < 8; ++nt) {
    v8f acc = {};
#pragma unroll
    for (int kt = 0; kt < 4; ++kt) {
      v16bf bb = frag_b(w1 + (nt * 16) * H_ + kt * 32, H_, lane);
      acc = wmma_bf16(af[kt], bb, acc);
    }
    const int col = nt * 16 + nl;
    const float bias = b1[a * H_ + col];
#pragma unroll
    for (int v = 0; v < 8; ++v) {
      float z = acc[v] + bias;
      z = z > 0.0f ? z : 0.01f * z;  // LeakyReLU(0.01)
      sZ[w][hh * 8 + v][col] = f2bf(z);
    }
  }
  __syncthreads();

  v8f accq = {};
  const bf16_t* w2 = W2T + a * (NACT_ * H_);
#pragma unroll
  for (int kt = 0; kt < 4; ++kt) {
    v16bf az = frag_a(&sZ[w][0][kt * 32], 136, lane);
    v16bf bb = frag_b(w2 + kt * 32, H_, lane);   // N = NACT = 16 (one tile)
    accq = wmma_bf16(az, bb, accq);
  }
  {
    const float bias = b2[a * NACT_ + nl];
#pragma unroll
    for (int v = 0; v < 8; ++v)
      sQ[w][hh * 8 + v][nl] = accq[v] + bias;
  }
  __syncthreads();

  if (lane < 16) {
    const int row  = lane;
    const int grow = r0 + row;
    const float* act = actions + ((size_t)a * B_ + grow) * NACT_;
    int amax = 0; float mv = act[0];
#pragma unroll
    for (int c = 1; c < NACT_; ++c) {
      float v = act[c];
      if (v > mv) { mv = v; amax = c; }   // first-max, matches jnp.argmax
    }
    qout[a * B_ + grow] = sQ[w][row][amax];
  }
}

// ---------------------------------------------------------------------------
extern "C" void kernel_launch(void* const* d_in, const int* in_sizes, int n_in,
                              void* d_out, int out_size, void* d_ws, size_t ws_size,
                              hipStream_t stream) {
  const float* unary   = (const float*)d_in[0];
  const int*   binary  = (const int*)d_in[1];
  const float* actions = (const float*)d_in[2];
  const float* We      = (const float*)d_in[3];
  const float* be      = (const float*)d_in[4];
  const float* Wrel    = (const float*)d_in[5];
  const float* Wroot   = (const float*)d_in[6];
  const float* brel    = (const float*)d_in[7];
  const float* W1      = (const float*)d_in[8];
  const float* b1      = (const float*)d_in[9];
  const float* W2      = (const float*)d_in[10];
  const float* b2      = (const float*)d_in[11];

  char* ws = (char*)d_ws;
  bf16_t* WrelT  = (bf16_t*)(ws + 0);        // 262144 B
  bf16_t* WrootT = (bf16_t*)(ws + 262144);   //  32768 B
  bf16_t* WeT    = (bf16_t*)(ws + 294912);   //  16384 B
  bf16_t* W1T    = (bf16_t*)(ws + 311296);   // 131072 B
  bf16_t* W2T    = (bf16_t*)(ws + 442368);   //  16384 B
  bf16_t* pooled = (bf16_t*)(ws + 458752);   // 524288 B  (total ~0.94 MB)

  relcrit_prep<<<896, 256, 0, stream>>>(We, Wrel, Wroot, W1, W2,
                                        WrelT, WrootT, WeT, W1T, W2T);
  relcrit_main<<<B_, 256, 0, stream>>>(unary, binary, be, brel,
                                       WeT, WrelT, WrootT, pooled);
  relcrit_heads<<<64, 256, 0, stream>>>(pooled, W1T, W2T, b1, b2, actions,
                                        (float*)d_out);
}
